// SelfAttention_88545045775046
// MI455X (gfx1250) — matrix operand order repaired and checked by reading, not compiled, not run on hardware
//
#include <hip/hip_runtime.h>
#include <hip/hip_bf16.h>

#define E_DIM 512
#define H_DIM 8
#define D_DIM 64
#define B_DIM 4
#define S_DIM 2048
#define NTOK  (B_DIM * S_DIM)   // 8192 tokens

typedef __bf16 v16bf __attribute__((ext_vector_type(16)));
typedef float  v8f   __attribute__((ext_vector_type(8)));

__device__ __forceinline__ unsigned short f32_to_bf16(float f) {
  unsigned int u = __float_as_uint(f);
  u += 0x7FFFu + ((u >> 16) & 1u);   // round-to-nearest-even
  return (unsigned short)(u >> 16);
}

__device__ __forceinline__ v8f wmma_bf16(v16bf a, v16bf b, v8f c) {
  // D = A(16x32 bf16) x B(32x16 bf16) + C(16x16 f32)
  return __builtin_amdgcn_wmma_f32_16x16x32_bf16(
      /*neg_a=*/false, a, /*neg_b=*/false, b,
      /*c_mod=*/(short)0, c, /*reuse_a=*/false, /*reuse_b=*/false);
}

union Frag16 {
  v16bf v;
  uint4 q[2];
  unsigned short s[16];
};

// B operand: lane = column (n = lane&15), K = (lane>>4)*16 .. +15 contiguous.
__device__ __forceinline__ v16bf load_b_frag(const unsigned short* p) {
  Frag16 f;
  f.q[0] = *(const uint4*)(p);
  f.q[1] = *(const uint4*)(p + 8);
  return f.v;
}

// A operand from bf16 row: elements 0..15 = row[kb..kb+15] contiguous
// (kb = (lane>>4)*16 within the 32-wide K step).
__device__ __forceinline__ v16bf load_a_frag_bf16(const unsigned short* row, int kb) {
  Frag16 f;
  f.q[0] = *(const uint4*)(row + kb);
  f.q[1] = *(const uint4*)(row + kb + 8);
  return f.v;
}

// ---------------------------------------------------------------------------
// Kernel 1: vectorized fp32 -> bf16 conversion (8 elements per thread).
// ---------------------------------------------------------------------------
__global__ void cvt_f32_bf16_kernel(const float* __restrict__ src,
                                    unsigned short* __restrict__ dst,
                                    int n8) {
  int i = blockIdx.x * blockDim.x + threadIdx.x;
  if (i < n8) {
    const float4* s = (const float4*)src + (size_t)i * 2;
    float4 a = s[0];
    float4 b = s[1];
    union { uint4 q; unsigned short u[8]; } o;
    o.u[0] = f32_to_bf16(a.x); o.u[1] = f32_to_bf16(a.y);
    o.u[2] = f32_to_bf16(a.z); o.u[3] = f32_to_bf16(a.w);
    o.u[4] = f32_to_bf16(b.x); o.u[5] = f32_to_bf16(b.y);
    o.u[6] = f32_to_bf16(b.z); o.u[7] = f32_to_bf16(b.w);
    ((uint4*)dst)[i] = o.q;
  }
}

// ---------------------------------------------------------------------------
// Kernel 2: projection GEMM  Y = Xbf (8192x512) * Wbf^T (512x512), bf16 in,
// f32 accumulate, bf16 out. One wave computes a 32x64 strip (8 accumulators):
// per K-step 2 A-frags + 4 B-frags feed 8 WMMAs. Software-pipelined (next
// K-step fragments preloaded before current WMMAs issue).
// v_transposed==0 : Y is bf16 [token, feature]
// v_transposed==1 : Y is bf16 vT[(b*E + feature)*S + s]  (for the P*V stage)
// ---------------------------------------------------------------------------
__global__ void proj_gemm_kernel(const unsigned short* __restrict__ Xbf,
                                 const unsigned short* __restrict__ Wbf,
                                 unsigned short* __restrict__ Y,
                                 int v_transposed) {
  const int wave = threadIdx.x >> 5;
  const int lane = threadIdx.x & 31;
  const int hl = lane >> 4;     // lane half
  const int ln = lane & 15;

  const int gid = blockIdx.x * 8 + wave;   // 2048 waves total
  const int rowGrp = gid >> 3;             // 256 groups of 32 rows
  const int colGrp = gid & 7;              // 8 groups of 64 columns

  const int kbA = hl * 16;
  const int kbB = hl * 16;

  const unsigned short* A0 = Xbf + (size_t)(rowGrp * 32 +  0 + ln) * E_DIM;
  const unsigned short* A1 = Xbf + (size_t)(rowGrp * 32 + 16 + ln) * E_DIM;
  const unsigned short* B0 = Wbf + (size_t)(colGrp * 64 +  0 + ln) * E_DIM + kbB;
  const unsigned short* B1 = Wbf + (size_t)(colGrp * 64 + 16 + ln) * E_DIM + kbB;
  const unsigned short* B2 = Wbf + (size_t)(colGrp * 64 + 32 + ln) * E_DIM + kbB;
  const unsigned short* B3 = Wbf + (size_t)(colGrp * 64 + 48 + ln) * E_DIM + kbB;

  v8f acc[2][4];
#pragma unroll
  for (int r = 0; r < 2; ++r)
#pragma unroll
    for (int c = 0; c < 4; ++c) acc[r][c] = (v8f){};

  // prologue: load K-step 0 fragments
  v16bf a0 = load_a_frag_bf16(A0, kbA);
  v16bf a1 = load_a_frag_bf16(A1, kbA);
  v16bf b0 = load_b_frag(B0);
  v16bf b1 = load_b_frag(B1);
  v16bf b2 = load_b_frag(B2);
  v16bf b3 = load_b_frag(B3);

  for (int kk = 0; kk < E_DIM - 32; kk += 32) {
    // preload next K-step while current WMMAs execute
    const int nk = kk + 32;
    v16bf na0 = load_a_frag_bf16(A0, nk + kbA);
    v16bf na1 = load_a_frag_bf16(A1, nk + kbA);
    v16bf nb0 = load_b_frag(B0 + nk);
    v16bf nb1 = load_b_frag(B1 + nk);
    v16bf nb2 = load_b_frag(B2 + nk);
    v16bf nb3 = load_b_frag(B3 + nk);

    acc[0][0] = wmma_bf16(a0, b0, acc[0][0]);
    acc[1][0] = wmma_bf16(a1, b0, acc[1][0]);
    acc[0][1] = wmma_bf16(a0, b1, acc[0][1]);
    acc[1][1] = wmma_bf16(a1, b1, acc[1][1]);
    acc[0][2] = wmma_bf16(a0, b2, acc[0][2]);
    acc[1][2] = wmma_bf16(a1, b2, acc[1][2]);
    acc[0][3] = wmma_bf16(a0, b3, acc[0][3]);
    acc[1][3] = wmma_bf16(a1, b3, acc[1][3]);

    a0 = na0; a1 = na1; b0 = nb0; b1 = nb1; b2 = nb2; b3 = nb3;
  }
  // epilogue: last K-step
  acc[0][0] = wmma_bf16(a0, b0, acc[0][0]);
  acc[1][0] = wmma_bf16(a1, b0, acc[1][0]);
  acc[0][1] = wmma_bf16(a0, b1, acc[0][1]);
  acc[1][1] = wmma_bf16(a1, b1, acc[1][1]);
  acc[0][2] = wmma_bf16(a0, b2, acc[0][2]);
  acc[1][2] = wmma_bf16(a1, b2, acc[1][2]);
  acc[0][3] = wmma_bf16(a0, b3, acc[0][3]);
  acc[1][3] = wmma_bf16(a1, b3, acc[1][3]);

  if (!v_transposed) {
#pragma unroll
    for (int rt = 0; rt < 2; ++rt)
#pragma unroll
      for (int c = 0; c < 4; ++c) {
        int col = colGrp * 64 + c * 16 + ln;
#pragma unroll
        for (int j = 0; j < 8; ++j) {
          int r = rowGrp * 32 + rt * 16 + j + 8 * hl;
          Y[(size_t)r * E_DIM + col] = f32_to_bf16(acc[rt][c][j]);
        }
      }
  } else {
#pragma unroll
    for (int rt = 0; rt < 2; ++rt)
#pragma unroll
      for (int c = 0; c < 4; ++c) {
        int col = colGrp * 64 + c * 16 + ln;
#pragma unroll
        for (int j = 0; j < 8; ++j) {
          int r = rowGrp * 32 + rt * 16 + j + 8 * hl;  // token
          int bb = r >> 11;                            // / S_DIM
          int ss = r & (S_DIM - 1);
          Y[((size_t)(bb * E_DIM + col)) * S_DIM + ss] = f32_to_bf16(acc[rt][c][j]);
        }
      }
  }
}

// ---------------------------------------------------------------------------
// Kernel 3: flash attention. One wave per (b, h, 16-query tile).
// K fragments are double-buffered across key chunks; V fragment loads are
// issued before the softmax so the exp/shfl VALU work hides their latency.
// ---------------------------------------------------------------------------
__global__ void attention_kernel(const unsigned short* __restrict__ qbf,
                                 const unsigned short* __restrict__ kbf,
                                 const unsigned short* __restrict__ vT,
                                 const int* __restrict__ mask,
                                 unsigned short* __restrict__ ctx) {
  __shared__ __align__(16) unsigned short Plds[8][16][32];  // 1KB per wave

  const int wave = threadIdx.x >> 5;
  const int lane = threadIdx.x & 31;
  const int hl = lane >> 4;
  const int ln = lane & 15;

  const int gid = blockIdx.x * 8 + wave;   // 4096 work units
  const int qt = gid & 127;                // S/16
  const int bh = gid >> 7;
  const int b = bh >> 3;
  const int h = bh & 7;

  const float scale = 0.04419417382415922f;  // 1/sqrt(512)
  const int kbA = hl * 16;
  const int kbB = hl * 16;

  // Q fragments (A operand), D=64 -> two K-steps of 32, loaded once.
  const int qrow = qt * 16 + ln;
  const unsigned short* qptr =
      qbf + ((size_t)(b * S_DIM + qrow)) * E_DIM + h * D_DIM;
  v16bf qa0 = load_a_frag_bf16(qptr, 0 + kbA);
  v16bf qa1 = load_a_frag_bf16(qptr, 32 + kbA);

  const unsigned short* kbase = kbf + ((size_t)b * S_DIM) * E_DIM + h * D_DIM;
  const unsigned short* vbase = vT + ((size_t)(b * E_DIM + h * D_DIM)) * S_DIM;
  const int* mrow = mask + b * S_DIM;

  v8f zero = {};
  v8f O[4] = {zero, zero, zero, zero};
  float m_i[8], l_i[8];
#pragma unroll
  for (int j = 0; j < 8; ++j) { m_i[j] = -3.0e38f; l_i[j] = 0.0f; }

  // prologue: K fragments for chunk 0 ([t*2 + ks] indexing)
  v16bf kc[4];
  {
    const unsigned short* kp0 = kbase + (size_t)(0 + ln) * E_DIM;
    const unsigned short* kp1 = kbase + (size_t)(16 + ln) * E_DIM;
    kc[0] = load_b_frag(kp0 + 0 + kbB);
    kc[1] = load_b_frag(kp0 + 32 + kbB);
    kc[2] = load_b_frag(kp1 + 0 + kbB);
    kc[3] = load_b_frag(kp1 + 32 + kbB);
  }

  for (int kk = 0; kk < S_DIM; kk += 32) {
    // preload next chunk's K fragments (uniform predicate)
    v16bf kn[4] = {kc[0], kc[1], kc[2], kc[3]};
    if (kk + 32 < S_DIM) {
      const unsigned short* kp0 = kbase + (size_t)(kk + 32 + ln) * E_DIM;
      const unsigned short* kp1 = kbase + (size_t)(kk + 48 + ln) * E_DIM;
      kn[0] = load_b_frag(kp0 + 0 + kbB);
      kn[1] = load_b_frag(kp0 + 32 + kbB);
      kn[2] = load_b_frag(kp1 + 0 + kbB);
      kn[3] = load_b_frag(kp1 + 32 + kbB);
    }

    // issue V fragment loads for the current chunk now; consumed after softmax
    v16bf vb[4];
#pragma unroll
    for (int t = 0; t < 4; ++t)
      vb[t] = load_b_frag(vbase + (size_t)(t * 16 + ln) * S_DIM + kk + kbB);

    // ---- energy: two 16x16 tiles over 32 keys -------------------------
    v8f e[2];
#pragma unroll
    for (int t = 0; t < 2; ++t) {
      v8f acc = zero;
      acc = wmma_bf16(qa0, kc[t * 2 + 0], acc);
      acc = wmma_bf16(qa1, kc[t * 2 + 1], acc);
      int mv = mrow[kk + t * 16 + ln];        // one key column per lane
#pragma unroll
      for (int j = 0; j < 8; ++j)
        acc[j] = (mv != 0) ? acc[j] * scale : -1.0e30f;
      e[t] = acc;
    }

    // ---- online softmax (row j lives in the 16-lane half) -------------
    float p0[8], p1[8];
#pragma unroll
    for (int j = 0; j < 8; ++j) {
      float rm = fmaxf(e[0][j], e[1][j]);
      rm = fmaxf(rm, __shfl_xor(rm, 1, 32));
      rm = fmaxf(rm, __shfl_xor(rm, 2, 32));
      rm = fmaxf(rm, __shfl_xor(rm, 4, 32));
      rm = fmaxf(rm, __shfl_xor(rm, 8, 32));
      float nm = fmaxf(m_i[j], rm);
      float corr = __expf(m_i[j] - nm);
      p0[j] = __expf(e[0][j] - nm);
      p1[j] = __expf(e[1][j] - nm);
      float rs = p0[j] + p1[j];
      rs += __shfl_xor(rs, 1, 32);
      rs += __shfl_xor(rs, 2, 32);
      rs += __shfl_xor(rs, 4, 32);
      rs += __shfl_xor(rs, 8, 32);
      l_i[j] = l_i[j] * corr + rs;
      m_i[j] = nm;
      O[0][j] *= corr; O[1][j] *= corr; O[2][j] *= corr; O[3][j] *= corr;
    }

    // ---- stage P through LDS: C-layout -> A-layout ---------------------
#pragma unroll
    for (int j = 0; j < 8; ++j) {
      int r = j + 8 * hl;
      Plds[wave][r][ln] = f32_to_bf16(p0[j]);
      Plds[wave][r][ln + 16] = f32_to_bf16(p1[j]);
    }
    asm volatile("s_wait_dscnt 0" ::: "memory");
    v16bf pa = load_a_frag_bf16(&Plds[wave][ln][0], kbA);

    // ---- O += P x V : 4 d-tiles ----------------------------------------
#pragma unroll
    for (int t = 0; t < 4; ++t) O[t] = wmma_bf16(pa, vb[t], O[t]);

    kc[0] = kn[0]; kc[1] = kn[1]; kc[2] = kn[2]; kc[3] = kn[3];
  }

  // ---- normalize, store context in bf16 [token, feature] ---------------
#pragma unroll
  for (int t = 0; t < 4; ++t) {
#pragma unroll
    for (int j = 0; j < 8; ++j) {
      int r = qt * 16 + j + 8 * hl;
      float o = O[t][j] / l_i[j];
      ctx[((size_t)(b * S_DIM + r)) * E_DIM + h * D_DIM + t * 16 + ln] =
          f32_to_bf16(o);
    }
  }
}

// ---------------------------------------------------------------------------
// Kernel 4: output projection  out = ctx (bf16) * Wo^T (bf16) + bo, fp32 out.
// Same 32x64-strip software-pipelined structure as the projection GEMM.
// ---------------------------------------------------------------------------
__global__ void out_gemm_kernel(const unsigned short* __restrict__ ctx,
                                const unsigned short* __restrict__ Wob,
                                const float* __restrict__ bo,
                                float* __restrict__ out) {
  const int wave = threadIdx.x >> 5;
  const int lane = threadIdx.x & 31;
  const int hl = lane >> 4;
  const int ln = lane & 15;

  const int gid = blockIdx.x * 8 + wave;   // 2048 waves
  const int rowGrp = gid >> 3;
  const int colGrp = gid & 7;

  const int kbA = hl * 16;
  const int kbB = hl * 16;

  const unsigned short* A0 = ctx + (size_t)(rowGrp * 32 +  0 + ln) * E_DIM;
  const unsigned short* A1 = ctx + (size_t)(rowGrp * 32 + 16 + ln) * E_DIM;
  const unsigned short* B0 = Wob + (size_t)(colGrp * 64 +  0 + ln) * E_DIM + kbB;
  const unsigned short* B1 = Wob + (size_t)(colGrp * 64 + 16 + ln) * E_DIM + kbB;
  const unsigned short* B2 = Wob + (size_t)(colGrp * 64 + 32 + ln) * E_DIM + kbB;
  const unsigned short* B3 = Wob + (size_t)(colGrp * 64 + 48 + ln) * E_DIM + kbB;

  v8f acc[2][4];
#pragma unroll
  for (int r = 0; r < 2; ++r)
#pragma unroll
    for (int c = 0; c < 4; ++c) acc[r][c] = (v8f){};

  v16bf a0 = load_a_frag_bf16(A0, kbA);
  v16bf a1 = load_a_frag_bf16(A1, kbA);
  v16bf b0 = load_b_frag(B0);
  v16bf b1 = load_b_frag(B1);
  v16bf b2 = load_b_frag(B2);
  v16bf b3 = load_b_frag(B3);

  for (int kk = 0; kk < E_DIM - 32; kk += 32) {
    const int nk = kk + 32;
    v16bf na0 = load_a_frag_bf16(A0, nk + kbA);
    v16bf na1 = load_a_frag_bf16(A1, nk + kbA);
    v16bf nb0 = load_b_frag(B0 + nk);
    v16bf nb1 = load_b_frag(B1 + nk);
    v16bf nb2 = load_b_frag(B2 + nk);
    v16bf nb3 = load_b_frag(B3 + nk);

    acc[0][0] = wmma_bf16(a0, b0, acc[0][0]);
    acc[1][0] = wmma_bf16(a1, b0, acc[1][0]);
    acc[0][1] = wmma_bf16(a0, b1, acc[0][1]);
    acc[1][1] = wmma_bf16(a1, b1, acc[1][1]);
    acc[0][2] = wmma_bf16(a0, b2, acc[0][2]);
    acc[1][2] = wmma_bf16(a1, b2, acc[1][2]);
    acc[0][3] = wmma_bf16(a0, b3, acc[0][3]);
    acc[1][3] = wmma_bf16(a1, b3, acc[1][3]);

    a0 = na0; a1 = na1; b0 = nb0; b1 = nb1; b2 = nb2; b3 = nb3;
  }
  acc[0][0] = wmma_bf16(a0, b0, acc[0][0]);
  acc[1][0] = wmma_bf16(a1, b0, acc[1][0]);
  acc[0][1] = wmma_bf16(a0, b1, acc[0][1]);
  acc[1][1] = wmma_bf16(a1, b1, acc[1][1]);
  acc[0][2] = wmma_bf16(a0, b2, acc[0][2]);
  acc[1][2] = wmma_bf16(a1, b2, acc[1][2]);
  acc[0][3] = wmma_bf16(a0, b3, acc[0][3]);
  acc[1][3] = wmma_bf16(a1, b3, acc[1][3]);

#pragma unroll
  for (int rt = 0; rt < 2; ++rt)
#pragma unroll
    for (int c = 0; c < 4; ++c) {
      int col = colGrp * 64 + c * 16 + ln;
      float bias = bo[col];
#pragma unroll
      for (int j = 0; j < 8; ++j) {
        int r = rowGrp * 32 + rt * 16 + j + 8 * hl;
        out[(size_t)r * E_DIM + col] = acc[rt][c][j] + bias;
      }
    }
}

// ---------------------------------------------------------------------------
extern "C" void kernel_launch(void* const* d_in, const int* in_sizes, int n_in,
                              void* d_out, int out_size, void* d_ws, size_t ws_size,
                              hipStream_t stream) {
  const float* values = (const float*)d_in[0];
  const float* keys   = (const float*)d_in[1];
  const float* query  = (const float*)d_in[2];
  const int*   mask   = (const int*)d_in[3];
  const float* Wv = (const float*)d_in[4];
  const float* Wk = (const float*)d_in[5];
  const float* Wq = (const float*)d_in[6];
  const float* Wo = (const float*)d_in[7];
  const float* bo = (const float*)d_in[8];
  float* out = (float*)d_out;

  // workspace layout (u16 elements):
  //   3 bf16 input copies + 4 bf16 weights + q + k + vT + ctx  (~58 MB)
  unsigned short* ws = (unsigned short*)d_ws;
  const size_t WSZ = (size_t)E_DIM * E_DIM;   // 262144
  const size_t XSZ = (size_t)NTOK * E_DIM;    // 4194304
  unsigned short* xvb = ws;
  unsigned short* xkb = xvb + XSZ;
  unsigned short* xqb = xkb + XSZ;
  unsigned short* wvb = xqb + XSZ;
  unsigned short* wkb = wvb + WSZ;
  unsigned short* wqb = wkb + WSZ;
  unsigned short* wob = wqb + WSZ;
  unsigned short* qbf = wob + WSZ;
  unsigned short* kbf = qbf + XSZ;
  unsigned short* vTb = kbf + XSZ;
  unsigned short* ctx = vTb + XSZ;

  // fp32 -> bf16 conversions (memory-bound, ~80 MB total traffic)
  const int X8 = (int)(XSZ / 8), W8 = (int)(WSZ / 8);
  cvt_f32_bf16_kernel<<<(X8 + 255) / 256, 256, 0, stream>>>(values, xvb, X8);
  cvt_f32_bf16_kernel<<<(X8 + 255) / 256, 256, 0, stream>>>(keys,   xkb, X8);
  cvt_f32_bf16_kernel<<<(X8 + 255) / 256, 256, 0, stream>>>(query,  xqb, X8);
  cvt_f32_bf16_kernel<<<(W8 + 255) / 256, 256, 0, stream>>>(Wv, wvb, W8);
  cvt_f32_bf16_kernel<<<(W8 + 255) / 256, 256, 0, stream>>>(Wk, wkb, W8);
  cvt_f32_bf16_kernel<<<(W8 + 255) / 256, 256, 0, stream>>>(Wq, wqb, W8);
  cvt_f32_bf16_kernel<<<(W8 + 255) / 256, 256, 0, stream>>>(Wo, wob, W8);

  // 2048 waves / 8 per block = 256 blocks (32x64 strip per wave)
  proj_gemm_kernel<<<256, 256, 0, stream>>>(xqb, wqb, qbf, 0);
  proj_gemm_kernel<<<256, 256, 0, stream>>>(xkb, wkb, kbf, 0);
  proj_gemm_kernel<<<256, 256, 0, stream>>>(xvb, wvb, vTb, 1);

  // 4096 (b,h,qtile) units / 8 waves per block = 512 blocks
  attention_kernel<<<512, 256, 0, stream>>>(qbf, kbf, vTb, mask, ctx);

  out_gemm_kernel<<<256, 256, 0, stream>>>(ctx, wob, bo, out);
}
